// ScaledDotProductAttention_65420941853258
// MI455X (gfx1250) — matrix operand order repaired and checked by reading, not compiled, not run on hardware
//
#include <hip/hip_runtime.h>

// Causal flash-attention for B=2,H=16,S=2048,D=64 (fp32 in/out) on gfx1250.
// WMMA f16 x f16 -> f32 accumulate, online softmax, ping-pong LDS pipeline,
// one workgroup barrier per 64-key tile.

#define S_LEN   2048
#define D_HEAD  64
#define N_BH    32          // B*H
#define BLOCK_M 64          // query rows per workgroup (16 per wave)
#define BLOCK_N 64          // keys per pipelined iteration
#define NWAVES  4
#define SCALE_F 0.125f      // 1/sqrt(64)

#define KS_STRIDE (D_HEAD + 8)    // 72 halves = 144B rows (16B aligned)
#define VS_STRIDE (D_HEAD + 4)    // 68 halves = 136B rows (8B aligned, 34 banks)
#define PS_STRIDE (BLOCK_N + 8)   // 72 halves

typedef __attribute__((ext_vector_type(16))) _Float16 v16h;
typedef __attribute__((ext_vector_type(8)))  _Float16 v8h;
typedef __attribute__((ext_vector_type(4)))  _Float16 v4h;
typedef __attribute__((ext_vector_type(8)))  float    v8f;

static __device__ __forceinline__ v8f wmma_f16(v16h a, v16h b, v8f c) {
    // v_wmma_f32_16x16x32_f16 : D = A*B + C
    return __builtin_amdgcn_wmma_f32_16x16x32_f16(
        false, a, false, b, (short)0, c, false, false);
}

static __device__ __forceinline__ v16h ld_bfrag(const _Float16* p) {
    // 16 contiguous halves, 16B-aligned -> two ds_load_b128
    v8h lo = *(const v8h*)(p);
    v8h hi = *(const v8h*)(p + 8);
    v16h r;
#pragma unroll
    for (int i = 0; i < 8; ++i) { r[i] = lo[i]; r[8 + i] = hi[i]; }
    return r;
}

__global__ __launch_bounds__(NWAVES * 32)
void fattn_causal_kernel(const float* __restrict__ Qg,
                         const float* __restrict__ Kg,
                         const float* __restrict__ Vg,
                         float* __restrict__ Og)
{
    __shared__ _Float16 Ks[2][BLOCK_N][KS_STRIDE];   // keys x d (row major)
    __shared__ _Float16 Vs[2][BLOCK_N][VS_STRIDE];   // keys x d (row major)
    __shared__ _Float16 Ps[NWAVES][16][PS_STRIDE];   // per-wave P tiles

    const int tid  = threadIdx.x;
    const int wave = tid >> 5;
    const int lane = tid & 31;
    const int lh   = lane >> 4;        // which 16-lane half
    const int l16  = lane & 15;

    const int    bh      = blockIdx.y;
    const size_t base    = (size_t)bh * S_LEN * D_HEAD;
    const int    q_block = blockIdx.x * BLOCK_M;
    const int    q0      = q_block + wave * 16;

    // ---- Q tile (16x64) -> two 16x32 A fragments (contiguous K layout:
    // lanes 0-15: K=[0..15]; lanes 16-31: K=[16..31])
    const float* qrow = Qg + base + (size_t)(q0 + l16) * D_HEAD;
    v16h aq[2];
#pragma unroll
    for (int f = 0; f < 2; ++f) {
        const int off = f * 32 + lh * 16;
#pragma unroll
        for (int i = 0; i < 8; ++i) {
            aq[f][i]     = (_Float16)qrow[off + i];
            aq[f][8 + i] = (_Float16)qrow[off + 8 + i];
        }
    }

    v8f acc[4] = {v8f{0}, v8f{0}, v8f{0}, v8f{0}};
    float mrow[8], lrow[8];
#pragma unroll
    for (int j = 0; j < 8; ++j) { mrow[j] = -__builtin_inff(); lrow[j] = 0.0f; }

    const int nIter = blockIdx.x + 1;            // causal: (q_block+64)/BLOCK_N

    // staging map: 128 threads -> 64 keys x 2 chunks of 32 d-elements
    const int s_key = tid >> 1;
    const int s_d   = (tid & 1) * 32;
    const float* kbase = Kg + base;
    const float* vbase = Vg + base;

    v8h kh[4];        // staged next K tile (32 halves, f16)
    v4h vh[8];        // staged next V tile (32 halves, f16)

    // ---- prologue: load tile 0 into registers, store to buffer 0
    {
        const float* kg = kbase + (size_t)s_key * D_HEAD + s_d;
        const float* vg = vbase + (size_t)s_key * D_HEAD + s_d;
#pragma unroll
        for (int c = 0; c < 4; ++c) {
            float4 a = ((const float4*)kg)[2 * c];
            float4 b = ((const float4*)kg)[2 * c + 1];
            kh[c] = v8h{(_Float16)a.x, (_Float16)a.y, (_Float16)a.z, (_Float16)a.w,
                        (_Float16)b.x, (_Float16)b.y, (_Float16)b.z, (_Float16)b.w};
        }
#pragma unroll
        for (int c = 0; c < 8; ++c) {
            float4 a = ((const float4*)vg)[c];
            vh[c] = v4h{(_Float16)a.x, (_Float16)a.y, (_Float16)a.z, (_Float16)a.w};
        }
    }
#pragma unroll
    for (int c = 0; c < 4; ++c) *(v8h*)&Ks[0][s_key][s_d + 8 * c] = kh[c];
#pragma unroll
    for (int c = 0; c < 8; ++c) *(v4h*)&Vs[0][s_key][s_d + 4 * c] = vh[c];
    __syncthreads();

    for (int it = 0; it < nIter; ++it) {
        const int n0  = it * BLOCK_N;
        const int cur = it & 1;
        const bool have_next = (it + 1) < nIter;

        if (it + 2 < nIter) {   // L2 prefetch two tiles ahead
            __builtin_prefetch(kbase + (size_t)((it + 2) * BLOCK_N + s_key) * D_HEAD + s_d, 0, 1);
            __builtin_prefetch(vbase + (size_t)((it + 2) * BLOCK_N + s_key) * D_HEAD + s_d, 0, 1);
        }
        // ---- issue next tile's global loads (latency overlaps compute below)
        if (have_next) {
            const float* kg = kbase + (size_t)(n0 + BLOCK_N + s_key) * D_HEAD + s_d;
            const float* vg = vbase + (size_t)(n0 + BLOCK_N + s_key) * D_HEAD + s_d;
#pragma unroll
            for (int c = 0; c < 4; ++c) {
                float4 a = ((const float4*)kg)[2 * c];
                float4 b = ((const float4*)kg)[2 * c + 1];
                kh[c] = v8h{(_Float16)a.x, (_Float16)a.y, (_Float16)a.z, (_Float16)a.w,
                            (_Float16)b.x, (_Float16)b.y, (_Float16)b.z, (_Float16)b.w};
            }
#pragma unroll
            for (int c = 0; c < 8; ++c) {
                float4 a = ((const float4*)vg)[c];
                vh[c] = v4h{(_Float16)a.x, (_Float16)a.y, (_Float16)a.z, (_Float16)a.w};
            }
        }

        // ---- S = Q K^T : four 16x16 score tiles (64 keys)
        v8f c4[4];
#pragma unroll
        for (int t = 0; t < 4; ++t) {
            v16h b0 = ld_bfrag(&Ks[cur][t * 16 + l16][lh * 16]);
            c4[t] = wmma_f16(aq[0], b0, v8f{0});
            v16h b1 = ld_bfrag(&Ks[cur][t * 16 + l16][32 + lh * 16]);
            c4[t] = wmma_f16(aq[1], b1, c4[t]);
        }

        // ---- mask + scale + online softmax; write P straight to LDS as f16.
        // Masking only ever needed on the diagonal tile (wave-uniform branch).
        const bool domask = (n0 + BLOCK_N - 1) > q0;
        float alpha[8];
#pragma unroll
        for (int j = 0; j < 8; ++j) {
            const int qr = q0 + lh * 8 + j;
            float s[4];
#pragma unroll
            for (int t = 0; t < 4; ++t) {
                s[t] = c4[t][j] * SCALE_F;
                if (domask && (n0 + t * 16 + l16 > qr)) s[t] = -__builtin_inff();
            }
            float mx = fmaxf(fmaxf(s[0], s[1]), fmaxf(s[2], s[3]));
#pragma unroll
            for (int m = 1; m < 16; m <<= 1)        // reduce within 16-lane group
                mx = fmaxf(mx, __shfl_xor(mx, m, 32));
            const float mnew = fmaxf(mrow[j], mx);
            alpha[j] = __expf(mrow[j] - mnew);
            float rs = 0.0f;
#pragma unroll
            for (int t = 0; t < 4; ++t) {
                const float p = __expf(s[t] - mnew);
                Ps[wave][lh * 8 + j][t * 16 + l16] = (_Float16)p;
                rs += p;
            }
#pragma unroll
            for (int m = 1; m < 16; m <<= 1)
                rs += __shfl_xor(rs, m, 32);
            lrow[j] = lrow[j] * alpha[j] + rs;
            mrow[j] = mnew;
        }
#pragma unroll
        for (int t = 0; t < 4; ++t)
#pragma unroll
            for (int j = 0; j < 8; ++j)
                acc[t][j] *= alpha[j];

        // P writes/reads are intra-wave; LDS ops of one wave execute in order,
        // so a compiler reordering fence is sufficient (no block barrier).
        asm volatile("" ::: "memory");

        v16h ap[2];
#pragma unroll
        for (int f = 0; f < 2; ++f) {
            const _Float16* pr = &Ps[wave][l16][f * 32 + lh * 16];
            v8h lo = *(const v8h*)(pr);
            v8h hi = *(const v8h*)(pr + 8);
#pragma unroll
            for (int i = 0; i < 8; ++i) { ap[f][i] = lo[i]; ap[f][8 + i] = hi[i]; }
        }

        // ---- O += P V : strided f16 gather of B frags from row-major Vs
#pragma unroll
        for (int t = 0; t < 4; ++t) {
#pragma unroll
            for (int f = 0; f < 2; ++f) {
                v16h bv;
#pragma unroll
                for (int i = 0; i < 16; ++i)
                    bv[i] = Vs[cur][f * 32 + lh * 16 + i][t * 16 + l16];
                acc[t] = wmma_f16(ap[f], bv, acc[t]);
            }
        }

        // ---- store next tile into the other buffer, single barrier per iter
        if (have_next) {
#pragma unroll
            for (int c = 0; c < 4; ++c) *(v8h*)&Ks[cur ^ 1][s_key][s_d + 8 * c] = kh[c];
#pragma unroll
            for (int c = 0; c < 8; ++c) *(v4h*)&Vs[cur ^ 1][s_key][s_d + 4 * c] = vh[c];
        }
        __syncthreads();
    }

    // ---- epilogue: O = acc / l   (C layout: row = j + 8*lh, col = t*16 + l16)
    float* orow = Og + base;
#pragma unroll
    for (int j = 0; j < 8; ++j) {
        const int qr  = q0 + lh * 8 + j;
        const float inv = 1.0f / lrow[j];
#pragma unroll
        for (int t = 0; t < 4; ++t)
            orow[(size_t)qr * D_HEAD + t * 16 + l16] = acc[t][j] * inv;
    }
}

extern "C" void kernel_launch(void* const* d_in, const int* in_sizes, int n_in,
                              void* d_out, int out_size, void* d_ws, size_t ws_size,
                              hipStream_t stream) {
    const float* Q = (const float*)d_in[0];
    const float* K = (const float*)d_in[1];
    const float* V = (const float*)d_in[2];
    // d_in[3] is the boolean causal mask; causality is applied analytically.
    float* O = (float*)d_out;

    dim3 grid(S_LEN / BLOCK_M, N_BH);
    dim3 block(NWAVES * 32);
    fattn_causal_kernel<<<grid, block, 0, stream>>>(Q, K, V, O);
}
